// BigramLanguageModel_46557445488808
// MI455X (gfx1250) — matrix operand order repaired and checked by reading, not compiled, not run on hardware
//
#include <hip/hip_runtime.h>

typedef __attribute__((ext_vector_type(16))) _Float16 v16h;
typedef __attribute__((ext_vector_type(8)))  _Float16 v8h;
typedef __attribute__((ext_vector_type(8)))  float    v8f;

#define Dm   64
#define Tm   32
#define Vm   65
#define Lm   4
#define FFm  256
#define NSEQ 4
#define ROWS 128        // NSEQ * Tm
#define NTHREADS 256    // 8 waves of 32

// ---------------- LDS layout (bytes) ----------------
#define OFF_XRES   0        // f32 [128][68]  residual stream
#define OFF_XN     34816    // f16 [128][72]  LN output (A operand)
#define OFF_Q      53248    // f16 [128][72]  Q, later attention-out
#define OFF_K      71680    // f16 [128][72]  K          (union with H)
#define OFF_VT     90112    // f16 [64][136]  V transposed (union with H)
#define OFF_SC     107520   // f32 [8][32][33] per-wave scores (union with H)
#define OFF_WEI    141312   // f16 [8][32][40] per-wave softmax weights
#define OFF_H      71680    // f16 [128][264] FF hidden (MLP phase only)
#define OFF_RED    161792   // f32 [128]      loss reduction
#define SMEM_BYTES 162304

// ---------------- workspace layout (halves): transposed f16 weights ----
#define WS_QKVO 0           // [L][4][64][64]  (Wq,Wk,Wv,Wo) as [n][k]
#define WS_W1T  65536       // [L][256][64]    W1^T
#define WS_W2T  131072      // [L][64][256]    W2^T
#define WS_WHT  196608      // [80][64]        Wh^T zero-padded to 80 rows
#define WS_TOTAL 201728

// ---------------- fragment helpers ----------------
__device__ __forceinline__ v8h zero_v8h() {
  v8h z;
#pragma unroll
  for (int i = 0; i < 8; ++i) z[i] = (_Float16)0.f;
  return z;
}
__device__ __forceinline__ v16h zero_v16h() {
  v16h z;
#pragma unroll
  for (int i = 0; i < 16; ++i) z[i] = (_Float16)0.f;
  return z;
}
__device__ __forceinline__ v8f zero_v8f() {
  v8f c;
#pragma unroll
  for (int i = 0; i < 8; ++i) c[i] = 0.f;
  return c;
}
// 16 contiguous halves (two b128 loads)
__device__ __forceinline__ v16h ldfrag16(const _Float16* p) {
  union { v16h v; v8h h[2]; } u;
  u.h[0] = *(const v8h*)p;
  u.h[1] = *(const v8h*)(p + 8);
  return u.v;
}
// A-matrix 16x32 f16 fragment from row-major [M][K] buffer.
// lanes 0-15: M=lane, K = {k0..k0+15}
// lanes 16-31: M=lane-16, K = {k0+16..k0+31}
__device__ __forceinline__ v16h ldA(const _Float16* buf, int stride, int m0, int k0, int lane) {
  int m = lane & 15, g = lane >> 4;
  const _Float16* p = buf + (m0 + m) * stride + k0 + 16 * g;
  union { v16h v; v8h h[2]; } u;
  u.h[0] = *(const v8h*)p;
  u.h[1] = *(const v8h*)(p + 8);
  return u.v;
}
// B-matrix 32x16 f16 fragment from N-major [N][K] buffer (transposed weight).
// lane = n + 16*g: VGPR j holds B[k0+16g+2j][n0+n] (lo), [.. +1] (hi)
__device__ __forceinline__ v16h ldB(const _Float16* buf, int stride, int n0, int k0, int lane) {
  int n = lane & 15, g = lane >> 4;
  return ldfrag16(buf + (n0 + n) * stride + k0 + 16 * g);
}
__device__ __forceinline__ v8f wmma_f16(v16h a, v16h b, v8f c) {
  // emits v_wmma_f32_16x16x32_f16
  return __builtin_amdgcn_wmma_f32_16x16x32_f16(false, a, false, b, (short)0, c, false, false);
}

// ---------------- prep: f32 weights -> transposed f16 in ws, zero loss ----
__global__ void prep_kernel(const float* __restrict__ Wq, const float* __restrict__ Wk,
                            const float* __restrict__ Wv, const float* __restrict__ Wo,
                            const float* __restrict__ W1, const float* __restrict__ W2,
                            const float* __restrict__ Wh,
                            _Float16* __restrict__ ws, float* __restrict__ loss) {
  int gid = blockIdx.x * blockDim.x + threadIdx.x;
  if (gid == 0) *loss = 0.f;
  if (gid < WS_W1T) {                       // qkvo: ((l*4+m)*64+n)*64+k
    int k = gid & 63, n = (gid >> 6) & 63, m = (gid >> 12) & 3, l = gid >> 14;
    const float* W = (m == 0) ? Wq : (m == 1) ? Wk : (m == 2) ? Wv : Wo;
    ws[gid] = (_Float16)W[(l * 64 + k) * 64 + n];
  } else if (gid < WS_W2T) {                // W1^T: ((l*256+n)*64+k)
    int t = gid - WS_W1T;
    int k = t & 63, n = (t >> 6) & 255, l = t >> 14;
    ws[gid] = (_Float16)W1[l * 16384 + k * 256 + n];
  } else if (gid < WS_WHT) {                // W2^T: ((l*64+n)*256+k)
    int t = gid - WS_W2T;
    int k = t & 255, n = (t >> 8) & 63, l = t >> 14;
    ws[gid] = (_Float16)W2[l * 16384 + k * 64 + n];
  } else if (gid < WS_TOTAL) {              // Wh^T padded: n*64+k, n<80
    int t = gid - WS_WHT;
    int k = t & 63, n = t >> 6;
    ws[gid] = (_Float16)((n < Vm) ? Wh[k * Vm + n] : 0.f);
  }
}

// ---------------- layernorm: xres(f32,stride 68) -> xn(f16,stride 72) ----
__device__ __forceinline__ void layernorm(const float* xres, _Float16* xn,
                                          const float* g, const float* b, int tid) {
  if (tid < ROWS) {
    const float* xr = xres + tid * 68;
    float mu = 0.f;
#pragma unroll
    for (int d = 0; d < Dm; ++d) mu += xr[d];
    mu *= (1.f / Dm);
    float var = 0.f;
#pragma unroll
    for (int d = 0; d < Dm; ++d) { float t = xr[d] - mu; var += t * t; }
    var *= (1.f / Dm);
    float inv = rsqrtf(var + 1e-5f);
    _Float16* xo = xn + tid * 72;
#pragma unroll
    for (int d = 0; d < Dm; ++d)
      xo[d] = (_Float16)((xr[d] - mu) * inv * g[d] + b[d]);
  }
}

// ---------------- fused transformer forward ----------------
__global__ void __launch_bounds__(NTHREADS, 1)
gpt_forward_kernel(const int* __restrict__ idx, const int* __restrict__ targets,
                   const float* __restrict__ tok_emb, const float* __restrict__ pos_emb,
                   const float* __restrict__ bo,
                   const float* __restrict__ ln1_g, const float* __restrict__ ln1_b,
                   const float* __restrict__ ln2_g, const float* __restrict__ ln2_b,
                   const float* __restrict__ b1, const float* __restrict__ b2,
                   const float* __restrict__ lnf_g, const float* __restrict__ lnf_b,
                   const float* __restrict__ bh,
                   const _Float16* __restrict__ ws,
                   float* __restrict__ out, float* __restrict__ loss, float inv_tokens) {
  extern __shared__ char smem[];
  float*    xres = (float*)(smem + OFF_XRES);
  _Float16* xn   = (_Float16*)(smem + OFF_XN);
  _Float16* qb   = (_Float16*)(smem + OFF_Q);
  _Float16* kb   = (_Float16*)(smem + OFF_K);
  _Float16* vT   = (_Float16*)(smem + OFF_VT);
  float*    sc   = (float*)(smem + OFF_SC);
  _Float16* wei  = (_Float16*)(smem + OFF_WEI);
  _Float16* hb   = (_Float16*)(smem + OFF_H);
  float*    red  = (float*)(smem + OFF_RED);

  const int tid  = threadIdx.x;
  const int lane = tid & 31;
  const int wave = tid >> 5;          // 0..7
  const int wg   = blockIdx.x;
  const int lnn  = lane & 15;         // N column within C/D tile
  const int lng  = lane >> 4;         // M-half within C/D tile

  // ---- embeddings: x = tok_emb[idx] + pos_emb ----
  {
    int r = tid >> 1, dh = (tid & 1) * 32;
    int gtok = wg * ROWS + r;
    int t = r & (Tm - 1);
    int tk = idx[gtok];
    const float* te = tok_emb + (size_t)tk * Dm;
    const float* pe = pos_emb + t * Dm;
    float* xr = xres + r * 68;
#pragma unroll
    for (int d = 0; d < 32; ++d) xr[dh + d] = te[dh + d] + pe[dh + d];
  }
  __syncthreads();

  for (int l = 0; l < Lm; ++l) {
    const _Float16* wqt = ws + WS_QKVO + (l * 4 + 0) * 4096;
    const _Float16* wkt = ws + WS_QKVO + (l * 4 + 1) * 4096;
    const _Float16* wvt = ws + WS_QKVO + (l * 4 + 2) * 4096;
    const _Float16* wot = ws + WS_QKVO + (l * 4 + 3) * 4096;
    const _Float16* w1t = ws + WS_W1T + l * 16384;
    const _Float16* w2t = ws + WS_W2T + l * 16384;

    // ---- LN1 ----
    layernorm(xres, xn, ln1_g + l * Dm, ln1_b + l * Dm, tid);
    __syncthreads();

    // ---- QKV projections: [128x64] @ [64x64], shared A fragment ----
#pragma unroll
    for (int i = 0; i < 4; ++i) {
      int tile = wave + 8 * i;                 // 32 tiles
      int m0 = (tile & 7) * 16, n0 = (tile >> 3) * 16;
      v8f cq = zero_v8f(), ck = zero_v8f(), cv = zero_v8f();
#pragma unroll
      for (int k0 = 0; k0 < Dm; k0 += 32) {
        v16h a = ldA(xn, 72, m0, k0, lane);
        cq = wmma_f16(a, ldB(wqt, 64, n0, k0, lane), cq);
        ck = wmma_f16(a, ldB(wkt, 64, n0, k0, lane), ck);
        cv = wmma_f16(a, ldB(wvt, 64, n0, k0, lane), cv);
      }
#pragma unroll
      for (int r = 0; r < 8; ++r) {
        int row = m0 + r + 8 * lng, col = n0 + lnn;
        qb[row * 72 + col] = (_Float16)cq[r];
        kb[row * 72 + col] = (_Float16)ck[r];
        vT[col * 136 + row] = (_Float16)cv[r];     // V stored transposed
      }
    }
    __syncthreads();

    // ---- attention: wave w -> seq w/2, heads {2*(w&1), 2*(w&1)+1} ----
    {
      int s = wave >> 1;
      int r0 = 32 * s;
      float*    scw = sc  + wave * (32 * 33);
      _Float16* ww  = wei + wave * (32 * 40);
      for (int hh = 0; hh < 2; ++hh) {
        int head = 2 * (wave & 1) + hh;
        int hc = head * 16;
        // scores = q @ k^T  (K=16 zero-padded to 32)
#pragma unroll
        for (int mi = 0; mi < 32; mi += 16) {
#pragma unroll
          for (int ni = 0; ni < 32; ni += 16) {
            union { v16h v; v8h h[2]; } ua;
            ua.h[0] = (lng == 0) ? *(const v8h*)(qb + (r0 + mi + lnn) * 72 + hc) : zero_v8h();
            ua.h[1] = (lng == 0) ? *(const v8h*)(qb + (r0 + mi + lnn) * 72 + hc + 8) : zero_v8h();
            v16h bfr;
            if (lng == 0) bfr = ldfrag16(kb + (r0 + ni + lnn) * 72 + hc);
            else          bfr = zero_v16h();
            v8f c = wmma_f16(ua.v, bfr, zero_v8f());
#pragma unroll
            for (int r = 0; r < 8; ++r)
              scw[(mi + r + 8 * lng) * 33 + ni + lnn] = c[r] * 0.125f; // * D^-0.5
          }
        }
        // causal softmax: lane i owns row i
        {
          int i = lane;
          float mx = -1e30f;
          for (int j = 0; j <= i; ++j) mx = fmaxf(mx, scw[i * 33 + j]);
          float sum = 0.f;
          for (int j = 0; j <= i; ++j) sum += __expf(scw[i * 33 + j] - mx);
          float inv = 1.f / sum;
          for (int j = 0; j < Tm; ++j)
            ww[i * 40 + j] = (_Float16)((j <= i) ? __expf(scw[i * 33 + j] - mx) * inv : 0.f);
        }
        // out_head = wei @ V  (K=32 exact); write in-place into q columns
#pragma unroll
        for (int mi = 0; mi < 32; mi += 16) {
          v16h a = ldA(ww, 40, mi, 0, lane);
          v16h b = ldfrag16(vT + (hc + lnn) * 136 + r0 + 16 * lng);
          v8f c = wmma_f16(a, b, zero_v8f());
#pragma unroll
          for (int r = 0; r < 8; ++r)
            qb[(r0 + mi + r + 8 * lng) * 72 + hc + lnn] = (_Float16)c[r];
        }
      }
    }
    __syncthreads();

    // ---- output projection + residual ----
#pragma unroll
    for (int i = 0; i < 4; ++i) {
      int tile = wave + 8 * i;
      int m0 = (tile & 7) * 16, n0 = (tile >> 3) * 16;
      v8f c = zero_v8f();
#pragma unroll
      for (int k0 = 0; k0 < Dm; k0 += 32)
        c = wmma_f16(ldA(qb, 72, m0, k0, lane), ldB(wot, 64, n0, k0, lane), c);
#pragma unroll
      for (int r = 0; r < 8; ++r) {
        int row = m0 + r + 8 * lng, col = n0 + lnn;
        xres[row * 68 + col] += c[r] + bo[l * Dm + col];
      }
    }
    __syncthreads();

    // ---- LN2 + MLP ----
    layernorm(xres, xn, ln2_g + l * Dm, ln2_b + l * Dm, tid);
    __syncthreads();

    // h = relu(xn @ W1 + b1): [128x64]@[64x256]
#pragma unroll
    for (int i = 0; i < 16; ++i) {
      int tile = wave + 8 * i;                 // 128 tiles
      int m0 = (tile & 7) * 16, n0 = (tile >> 3) * 16;
      v8f c = zero_v8f();
#pragma unroll
      for (int k0 = 0; k0 < Dm; k0 += 32)
        c = wmma_f16(ldA(xn, 72, m0, k0, lane), ldB(w1t, 64, n0, k0, lane), c);
#pragma unroll
      for (int r = 0; r < 8; ++r) {
        int row = m0 + r + 8 * lng, col = n0 + lnn;
        hb[row * 264 + col] = (_Float16)fmaxf(0.f, c[r] + b1[l * FFm + col]);
      }
    }
    __syncthreads();

    // x += h @ W2 + b2: [128x256]@[256x64]
#pragma unroll
    for (int i = 0; i < 4; ++i) {
      int tile = wave + 8 * i;
      int m0 = (tile & 7) * 16, n0 = (tile >> 3) * 16;
      v8f c = zero_v8f();
#pragma unroll
      for (int k0 = 0; k0 < FFm; k0 += 32)
        c = wmma_f16(ldA(hb, 264, m0, k0, lane), ldB(w2t, 256, n0, k0, lane), c);
#pragma unroll
      for (int r = 0; r < 8; ++r) {
        int row = m0 + r + 8 * lng, col = n0 + lnn;
        xres[row * 68 + col] += c[r] + b2[l * Dm + col];
      }
    }
    __syncthreads();
  }

  // ---- final LN + LM head ----
  layernorm(xres, xn, lnf_g, lnf_b, tid);
  __syncthreads();

  const _Float16* wht = ws + WS_WHT;
#pragma unroll
  for (int i = 0; i < 5; ++i) {
    int tile = wave + 8 * i;                   // 40 tiles (N padded to 80)
    int m0 = (tile & 7) * 16, n0 = (tile >> 3) * 16;
    v8f c = zero_v8f();
#pragma unroll
    for (int k0 = 0; k0 < Dm; k0 += 32)
      c = wmma_f16(ldA(xn, 72, m0, k0, lane), ldB(wht, 64, n0, k0, lane), c);
#pragma unroll
    for (int r = 0; r < 8; ++r) {
      int row = m0 + r + 8 * lng, col = n0 + lnn;
      if (col < Vm) {
        float val = c[r] + bh[col];
        out[(size_t)(wg * ROWS + row) * Vm + col] = val;  // logits
        xres[row * 68 + col] = val;                       // for loss
      }
    }
  }
  __syncthreads();

  // ---- cross-entropy loss contribution ----
  float li = 0.f;
  if (tid < ROWS) {
    const float* lg = xres + tid * 68;
    float mx = -1e30f;
    for (int j = 0; j < Vm; ++j) mx = fmaxf(mx, lg[j]);
    float sum = 0.f;
    for (int j = 0; j < Vm; ++j) sum += __expf(lg[j] - mx);
    int tgt = targets[wg * ROWS + tid];
    li = mx + __logf(sum) - lg[tgt];
    red[tid] = li;
  }
  __syncthreads();
  for (int off = 64; off > 0; off >>= 1) {
    if (tid < off) red[tid] += red[tid + off];
    __syncthreads();
  }
  if (tid == 0) atomicAdd(loss, red[0] * inv_tokens);
}

// ---------------- host launch ----------------
extern "C" void kernel_launch(void* const* d_in, const int* in_sizes, int n_in,
                              void* d_out, int out_size, void* d_ws, size_t ws_size,
                              hipStream_t stream) {
  (void)n_in; (void)ws_size; (void)out_size;
  const int*   idx     = (const int*)d_in[0];
  const int*   targets = (const int*)d_in[1];
  const float* tok_emb = (const float*)d_in[2];
  const float* pos_emb = (const float*)d_in[3];
  const float* Wq      = (const float*)d_in[4];
  const float* Wk      = (const float*)d_in[5];
  const float* Wv      = (const float*)d_in[6];
  const float* Wo      = (const float*)d_in[7];
  const float* bo      = (const float*)d_in[8];
  const float* ln1_g   = (const float*)d_in[9];
  const float* ln1_b   = (const float*)d_in[10];
  const float* ln2_g   = (const float*)d_in[11];
  const float* ln2_b   = (const float*)d_in[12];
  const float* W1      = (const float*)d_in[13];
  const float* b1      = (const float*)d_in[14];
  const float* W2      = (const float*)d_in[15];
  const float* b2      = (const float*)d_in[16];
  const float* lnf_g   = (const float*)d_in[17];
  const float* lnf_b   = (const float*)d_in[18];
  const float* Wh      = (const float*)d_in[19];
  const float* bh      = (const float*)d_in[20];

  _Float16* wsh  = (_Float16*)d_ws;
  float*    out  = (float*)d_out;
  const int nTok = in_sizes[0];              // B*T = 262144
  float*    loss = out + (size_t)nTok * Vm;  // loss slot after logits
  const int nB   = nTok / Tm;
  const int grid = nB / NSEQ;

  prep_kernel<<<(WS_TOTAL + 255) / 256, 256, 0, stream>>>(Wq, Wk, Wv, Wo, W1, W2, Wh, wsh, loss);
  gpt_forward_kernel<<<grid, NTHREADS, SMEM_BYTES, stream>>>(
      idx, targets, tok_emb, pos_emb, bo, ln1_g, ln1_b, ln2_g, ln2_b,
      b1, b2, lnf_g, lnf_b, bh, wsh, out, loss, 1.f / (float)nTok);
}
